// BenchCADecoder_31035433681594
// MI455X (gfx1250) — matrix operand order repaired and checked by reading, not compiled, not run on hardware
//
#include <hip/hip_runtime.h>
#include <math.h>

// ---------------- problem constants (from reference) ----------------
#define D_DIM  512
#define T_DIM  2048
#define B_DIM  4
#define R_DIM  8
#define N_STEPS 5
#define V_DIM  32000
#define ROWS   (B_DIM * T_DIM)      // 8192
#define HID    (2 * D_DIM)          // 1024
#define KIN    (3 * D_DIM)          // 1536
#define GATE_SCALE 0.001f
#define LN_EPS 1e-5f

// ---------------- CDNA5 WMMA types ----------------
typedef __attribute__((ext_vector_type(16))) __bf16 v16bf;
typedef __attribute__((ext_vector_type(8)))  __bf16 bf16x8;
typedef __attribute__((ext_vector_type(8)))  float  v8f;

__device__ __forceinline__ unsigned bf_rne(float f) {
  unsigned u = __float_as_uint(f);
  return (u + 0x7FFFu + ((u >> 16) & 1u)) >> 16;   // 16-bit result
}
__device__ __forceinline__ __bf16 f2bf(float f) {
  union { unsigned short u; __bf16 b; } cv;
  cv.u = (unsigned short)bf_rne(f);
  return cv.b;
}

// Async memory -> LDS DMA (CDNA5).  Per-lane: LDS[ldsoff] = MEM[gaddr] (16B).
// Tracked by ASYNCcnt, which the compiler never uses itself.
__device__ __forceinline__ void async_copy_b128(unsigned ldsoff, const void* gaddr) {
  asm volatile("global_load_async_to_lds_b128 %0, %1, off"
               :: "v"(ldsoff), "v"((unsigned long long)(size_t)gaddr)
               : "memory");
}
__device__ __forceinline__ void async_join() {
  asm volatile("s_wait_asynccnt 0x0" ::: "memory");
}

// ====================================================================
// Weight transpose + convert: src f32 [K][N] (+z*stride) -> dst bf16 [N][K]
// ====================================================================
__global__ __launch_bounds__(256)
void transpose_cvt_kernel(const float* __restrict__ src, __bf16* __restrict__ dst,
                          int K, int N, size_t zstride) {
  __shared__ float tile[32][33];
  src += (size_t)blockIdx.z * zstride;
  dst += (size_t)blockIdx.z * zstride;
  int nb = blockIdx.x * 32, kb = blockIdx.y * 32;
  int tx = threadIdx.x & 31, ty = threadIdx.x >> 5;   // ty in [0,8)
#pragma unroll
  for (int i = ty; i < 32; i += 8)
    tile[i][tx] = src[(size_t)(kb + i) * N + nb + tx];
  __syncthreads();
#pragma unroll
  for (int i = ty; i < 32; i += 8)
    dst[(size_t)(nb + i) * K + kb + tx] = f2bf(tile[tx][i]);
}

// ====================================================================
// Embedding: x[b,t,:] = embed[tokens[b,t]] + pos_embed[t]  (f32 + bf16 shadow)
// ====================================================================
__global__ void embed_kernel(const int* __restrict__ tokens,
                             const float* __restrict__ embed,
                             const float* __restrict__ pos,
                             float* __restrict__ x,
                             __bf16* __restrict__ xb) {
  int idx = blockIdx.x * blockDim.x + threadIdx.x;
  if (idx >= ROWS * D_DIM) return;
  int d   = idx & (D_DIM - 1);
  int row = idx >> 9;
  int t   = row & (T_DIM - 1);
  int tok = tokens[row];
  float v = embed[(size_t)tok * D_DIM + d] + pos[(size_t)t * D_DIM + d];
  x[idx]  = v;
  xb[idx] = f2bf(v);
}

// ====================================================================
// Rule selector
// ====================================================================
__global__ void sel_logits_kernel(const float* __restrict__ gate,
                                  const float* __restrict__ sw,
                                  const float* __restrict__ sb,
                                  float* __restrict__ rw) {
  int idx = blockIdx.x * blockDim.x + threadIdx.x;
  if (idx >= T_DIM * R_DIM) return;
  int t = idx >> 3, r = idx & 7;
  float acc = sb[r];
  const float* g = gate + (size_t)t * D_DIM;
  for (int d = 0; d < D_DIM; ++d)
    acc += g[d] * GATE_SCALE * sw[(size_t)d * R_DIM + r];
  rw[idx] = acc;
}

__global__ void sel_softmax_kernel(float* __restrict__ rw) {
  int t = blockIdx.x * blockDim.x + threadIdx.x;
  if (t >= T_DIM) return;
  float* p = rw + (size_t)t * R_DIM;
  float mx = p[0];
  for (int i = 1; i < R_DIM; ++i) mx = fmaxf(mx, p[i]);
  float e[R_DIM], s = 0.f;
  for (int i = 0; i < R_DIM; ++i) { e[i] = expf(p[i] - mx); s += e[i]; }
  float inv = 1.f / s;
  for (int i = 0; i < R_DIM; ++i) p[i] = e[i] * inv;
}

// ====================================================================
// Fused (x + evolved) -> LayerNorm.
// Writes f32 to outf (if non-null) and bf16 to outb (if non-null).
// ====================================================================
__global__ __launch_bounds__(256)
void addln_kernel(const float* __restrict__ xin,
                  const float* __restrict__ ev,
                  const float* __restrict__ g,
                  const float* __restrict__ bt,
                  float* __restrict__ outf,
                  __bf16* __restrict__ outb) {
  int row = blockIdx.x;
  int tid = threadIdx.x;
  __shared__ float red[256];
  __shared__ float s_mean, s_rstd;
  size_t base = (size_t)row * D_DIM;

  float v0 = xin[base + tid];
  float v1 = xin[base + tid + 256];
  if (ev) { v0 += ev[base + tid]; v1 += ev[base + tid + 256]; }

  red[tid] = v0 + v1;
  __syncthreads();
  for (int s = 128; s > 0; s >>= 1) {
    if (tid < s) red[tid] += red[tid + s];
    __syncthreads();
  }
  if (tid == 0) s_mean = red[0] * (1.f / (float)D_DIM);
  __syncthreads();
  float m  = s_mean;
  float d0 = v0 - m, d1 = v1 - m;
  red[tid] = d0 * d0 + d1 * d1;
  __syncthreads();
  for (int s = 128; s > 0; s >>= 1) {
    if (tid < s) red[tid] += red[tid + s];
    __syncthreads();
  }
  if (tid == 0) s_rstd = rsqrtf(red[0] * (1.f / (float)D_DIM) + LN_EPS);
  __syncthreads();
  float rs = s_rstd;
  float r0 = d0 * rs * g[tid]       + bt[tid];
  float r1 = d1 * rs * g[tid + 256] + bt[tid + 256];
  if (outf) {
    outf[base + tid]       = r0;
    outf[base + tid + 256] = r1;
  }
  if (outb) {
    outb[base + tid]       = f2bf(r0);
    outb[base + tid + 256] = f2bf(r1);
  }
}

// ====================================================================
// bf16 WMMA GEMM, double-buffered LDS, async DMA staging.
//   A (bf16): ASRC 0 = plain [M][K] row-major (lda)
//             ASRC 1 = implicit nb = concat(xb, roll(+1), roll(-1)) from
//                      xb[B,T,D] bf16; K = 3*D, roll along T with wrap.
//   B: bf16 [N][K] (pre-transposed), row stride = K
//   EPI 0: C f32 = val                            (head -> d_out)
//   EPI 1: C bf16 = gelu(val)                     (rule hidden H)
//   EPI 2: C f32 += rowscale[(row%T)*R] * val     (evolved, rules 1..R-1)
//   EPI 3: C f32  = rowscale[(row%T)*R] * val     (evolved, rule 0)
// Tile 128x128x32, 256 threads = 8 wave32, 2x4 WMMA 16x16 tiles per wave.
// Staging: 4x global_load_async_to_lds_b128 per thread per K-slab.
// ====================================================================
#define BM 128
#define BN 128
#define BK 32

template <int ASRC, int EPI>
__global__ __launch_bounds__(256)
void gemm_kernel(const __bf16* __restrict__ Amat,
                 const __bf16* __restrict__ Bmat,
                 const float* __restrict__ bias,
                 const float* __restrict__ rowscale,
                 void* __restrict__ Cptr,
                 int K, int lda, int ldc) {
  __shared__ __attribute__((aligned(16))) __bf16 As[2][BM * BK];
  __shared__ __attribute__((aligned(16))) __bf16 Bs[2][BN * BK];

  const int tid  = threadIdx.x;
  const int lane = tid & 31;
  const int wave = tid >> 5;
  const int wm   = wave & 3;     // 4 row groups of 32 rows
  const int wn   = wave >> 2;    // 2 col groups of 64 cols
  const int r16  = lane & 15;
  const int hi   = lane >> 4;

  const int blockRow = blockIdx.y * BM;
  const int blockCol = blockIdx.x * BN;

  v8f acc[2][4];
#pragma unroll
  for (int i = 0; i < 2; ++i)
#pragma unroll
    for (int j = 0; j < 4; ++j)
#pragma unroll
      for (int v = 0; v < 8; ++v) acc[i][j][v] = 0.f;

  // Issue async DMA for one 32-deep K slab into LDS buffer `buf`.
  auto stage = [&](int k0, int buf) {
#pragma unroll
    for (int j = 0; j < 2; ++j) {          // A: 512 x 16B chunks
      int i = tid + j * 256;
      int row = i >> 2, part = i & 3;
      int gk = k0 + part * 8;
      const __bf16* gp;
      if (ASRC == 1) {
        int gr  = blockRow + row;
        int b   = gr >> 11;                // T_DIM == 2048
        int t   = gr & (T_DIM - 1);
        int seg = gk >> 9;                 // D_DIM == 512
        int d   = gk & (D_DIM - 1);
        int t2  = (seg == 0) ? t
                : (seg == 1) ? ((t + T_DIM - 1) & (T_DIM - 1))
                             : ((t + 1) & (T_DIM - 1));
        gp = &Amat[(((size_t)(b << 11) + t2) << 9) + d];
      } else {
        gp = &Amat[(size_t)(blockRow + row) * lda + gk];
      }
      async_copy_b128((unsigned)(size_t)&As[buf][row * BK + part * 8], gp);
    }
#pragma unroll
    for (int j = 0; j < 2; ++j) {          // B: 512 x 16B chunks
      int i = tid + j * 256;
      int col = i >> 2, part = i & 3;
      const __bf16* gp = &Bmat[(size_t)(blockCol + col) * K + k0 + part * 8];
      async_copy_b128((unsigned)(size_t)&Bs[buf][col * BK + part * 8], gp);
    }
  };

  const int S = K / BK;
  stage(0, 0);
  async_join();
  __syncthreads();

  for (int s = 0; s < S; ++s) {
    int cur = s & 1, nxt = cur ^ 1;
    if (s + 1 < S) stage((s + 1) * BK, nxt);   // DMA overlaps WMMA below

    // ---- WMMA over the current 32-deep K slab ----
    v16bf afrag[2], bfrag[4];
#pragma unroll
    for (int mi = 0; mi < 2; ++mi) {
      int row = wm * 32 + mi * 16 + r16;
      bf16x8 lo = *(const bf16x8*)&As[cur][row * BK + hi * 8];
      bf16x8 hh = *(const bf16x8*)&As[cur][row * BK + 16 + hi * 8];
#pragma unroll
      for (int j = 0; j < 8; ++j) { afrag[mi][j] = lo[j]; afrag[mi][8 + j] = hh[j]; }
    }
#pragma unroll
    for (int ni = 0; ni < 4; ++ni) {
      int col = wn * 64 + ni * 16 + r16;
      bf16x8 lo = *(const bf16x8*)&Bs[cur][col * BK + hi * 8];
      bf16x8 hh = *(const bf16x8*)&Bs[cur][col * BK + 16 + hi * 8];
#pragma unroll
      for (int j = 0; j < 8; ++j) { bfrag[ni][j] = lo[j]; bfrag[ni][8 + j] = hh[j]; }
    }
#pragma unroll
    for (int mi = 0; mi < 2; ++mi)
#pragma unroll
      for (int ni = 0; ni < 4; ++ni)
        acc[mi][ni] = __builtin_amdgcn_wmma_f32_16x16x32_bf16(
            false, afrag[mi], false, bfrag[ni], (short)0, acc[mi][ni],
            false, false);

    async_join();       // our DMA for nxt complete (no-op on last iter)
    __syncthreads();    // publish across waves
  }

  // ---- epilogue ----
  float*  Cf = (float*)Cptr;
  __bf16* Cb = (__bf16*)Cptr;
#pragma unroll
  for (int mi = 0; mi < 2; ++mi) {
#pragma unroll
    for (int ni = 0; ni < 4; ++ni) {
      int gcol = blockCol + wn * 64 + ni * 16 + r16;
      float bval = (bias != nullptr) ? bias[gcol] : 0.f;
#pragma unroll
      for (int v = 0; v < 8; ++v) {
        int grow  = blockRow + wm * 32 + mi * 16 + hi * 8 + v;
        float val = acc[mi][ni][v] + bval;
        size_t idx = (size_t)grow * ldc + gcol;
        if (EPI == 1) {
          val = 0.5f * val * (1.0f + erff(val * 0.70710678118654752f));
          Cb[idx] = f2bf(val);
        } else if (EPI == 2) {
          float sc = rowscale[(size_t)(grow & (T_DIM - 1)) * R_DIM];
          Cf[idx] += sc * val;
        } else if (EPI == 3) {
          float sc = rowscale[(size_t)(grow & (T_DIM - 1)) * R_DIM];
          Cf[idx] = sc * val;
        } else {
          Cf[idx] = val;
        }
      }
    }
  }
}

// ====================================================================
// launch
// ====================================================================
extern "C" void kernel_launch(void* const* d_in, const int* in_sizes, int n_in,
                              void* d_out, int out_size, void* d_ws, size_t ws_size,
                              hipStream_t stream) {
  (void)in_sizes; (void)n_in; (void)out_size; (void)ws_size;

  const int*   tokens = (const int*)d_in[0];
  const float* gate   = (const float*)d_in[1];
  const float* embedW = (const float*)d_in[2];
  const float* posW   = (const float*)d_in[3];
  const float* w1     = (const float*)d_in[4];   // [R,3D,2D]
  const float* b1     = (const float*)d_in[5];   // [R,2D]
  const float* w2     = (const float*)d_in[6];   // [R,2D,D]
  const float* b2     = (const float*)d_in[7];   // [R,D]
  const float* sel_w  = (const float*)d_in[8];
  const float* sel_b  = (const float*)d_in[9];
  const float* norm_g = (const float*)d_in[10];  // [STEPS,D]
  const float* norm_b = (const float*)d_in[11];
  const float* lnf_g  = (const float*)d_in[12];
  const float* lnf_b  = (const float*)d_in[13];
  const float* head_w = (const float*)d_in[14];  // [D,V]
  float* out = (float*)d_out;

  // ---- workspace carve-up ----
  char* wp = (char*)d_ws;
  auto alloc = [&](size_t bytes) -> void* {
    void* q = (void*)wp;
    wp += (bytes + 255) & ~(size_t)255;
    return q;
  };
  float*  x      = (float*)alloc((size_t)ROWS * D_DIM * 4);
  float*  ev     = (float*)alloc((size_t)ROWS * D_DIM * 4);
  float*  rw     = (float*)alloc((size_t)T_DIM * R_DIM * 4);
  __bf16* xb     = (__bf16*)alloc((size_t)ROWS * D_DIM * 2);
  __bf16* Hbf    = (__bf16*)alloc((size_t)ROWS * HID * 2);
  __bf16* lnxbf  = (__bf16*)alloc((size_t)ROWS * D_DIM * 2);
  __bf16* w1t    = (__bf16*)alloc((size_t)R_DIM * KIN * HID * 2);
  __bf16* w2t    = (__bf16*)alloc((size_t)R_DIM * HID * D_DIM * 2);
  __bf16* headt  = (__bf16*)alloc((size_t)D_DIM * V_DIM * 2);

  // ---- one-time weight transpose/convert to bf16 [N][K] ----
  transpose_cvt_kernel<<<dim3(HID / 32, KIN / 32, R_DIM), 256, 0, stream>>>(
      w1, w1t, KIN, HID, (size_t)KIN * HID);
  transpose_cvt_kernel<<<dim3(D_DIM / 32, HID / 32, R_DIM), 256, 0, stream>>>(
      w2, w2t, HID, D_DIM, (size_t)HID * D_DIM);
  transpose_cvt_kernel<<<dim3(V_DIM / 32, D_DIM / 32, 1), 256, 0, stream>>>(
      head_w, headt, D_DIM, V_DIM, 0);

  // ---- embeddings + selector ----
  {
    int n = ROWS * D_DIM;
    embed_kernel<<<(n + 255) / 256, 256, 0, stream>>>(tokens, embedW, posW, x, xb);
  }
  {
    int n = T_DIM * R_DIM;
    sel_logits_kernel<<<(n + 255) / 256, 256, 0, stream>>>(gate, sel_w, sel_b, rw);
    sel_softmax_kernel<<<(T_DIM + 255) / 256, 256, 0, stream>>>(rw);
  }

  // ---- CA steps ----
  dim3 g1(HID / BN,   ROWS / BM);   // (8, 64)
  dim3 g2(D_DIM / BN, ROWS / BM);   // (4, 64)
  for (int step = 0; step < N_STEPS; ++step) {
    for (int r = 0; r < R_DIM; ++r) {
      // Hbf = gelu(nb @ w1_r + b1_r)  [8192 x 1024], K = 1536
      gemm_kernel<1, 1><<<g1, 256, 0, stream>>>(
          xb, w1t + (size_t)r * KIN * HID, b1 + (size_t)r * HID,
          nullptr, Hbf, KIN, /*lda*/0, /*ldc*/HID);
      // evolved (r==0: =, else: +=)  rw[t,r] * (Hbf @ w2_r + b2_r)
      if (r == 0)
        gemm_kernel<0, 3><<<g2, 256, 0, stream>>>(
            Hbf, w2t + (size_t)r * HID * D_DIM, b2 + (size_t)r * D_DIM,
            rw + r, ev, HID, /*lda*/HID, /*ldc*/D_DIM);
      else
        gemm_kernel<0, 2><<<g2, 256, 0, stream>>>(
            Hbf, w2t + (size_t)r * HID * D_DIM, b2 + (size_t)r * D_DIM,
            rw + r, ev, HID, /*lda*/HID, /*ldc*/D_DIM);
    }
    // x = LN(x + evolved)  (f32 + bf16 shadow)
    addln_kernel<<<ROWS, 256, 0, stream>>>(x, ev,
        norm_g + (size_t)step * D_DIM, norm_b + (size_t)step * D_DIM, x, xb);
  }

  // ---- final LN (bf16 only) and head projection ----
  addln_kernel<<<ROWS, 256, 0, stream>>>(x, nullptr, lnf_g, lnf_b, nullptr, lnxbf);

  dim3 gh(V_DIM / BN, ROWS / BM);   // (250, 64)
  gemm_kernel<0, 0><<<gh, 256, 0, stream>>>(
      lnxbf, headt, nullptr, nullptr, out, D_DIM,
      /*lda*/D_DIM, /*ldc*/V_DIM);
}
